// LocalAttention_MH_50508815401485
// MI455X (gfx1250) — matrix operand order repaired and checked by reading, not compiled, not run on hardware
//
#include <hip/hip_runtime.h>
#include <hip/hip_bf16.h>

// ---- CDNA5 WMMA / TDM types ----
typedef __attribute__((ext_vector_type(16))) __bf16      v16bf;
typedef __attribute__((ext_vector_type(8)))  float       v8f;
typedef __attribute__((ext_vector_type(4)))  unsigned    v4u;
typedef __attribute__((ext_vector_type(8)))  int         v8i;
typedef __attribute__((ext_vector_type(4)))  int         v4i;

// 32-bit LDS byte offset of a shared-memory pointer (generic -> addrspace(3) -> int)
__device__ __forceinline__ unsigned lds_off(const void* p)
{
    return (unsigned)(unsigned long long)(const __attribute__((address_space(3))) void*)p;
}

// ---- Tensor Data Mover: async 2D bf16 tile load Global->LDS (CDNA5 ISA ch.8, D# groups) ----
// Loads tile_h rows x tile_w elements (2B each), source row stride = row_stride elements,
// packed contiguously into LDS at lds_byte_off. Issue from ONE wave only (EXEC ignored).
// This toolchain's builtin is the 6-arg form: (v4u, v8i, v4i, v4i, v8i, i32 cpol).
__device__ __forceinline__ void tdm_load_tile_bf16(const __bf16* gsrc, unsigned lds_byte_off,
                                                   unsigned tile_w, unsigned tile_h,
                                                   unsigned long long row_stride)
{
    const unsigned long long ga = (unsigned long long)gsrc;
    v4u g0;
    g0.x = 1u;                                            // count=1, user desc, no gather
    g0.y = lds_byte_off;                                  // lds_addr [63:32]
    g0.z = (unsigned)ga;                                  // global_addr lo
    g0.w = (unsigned)((ga >> 32) & 0x01ffffffu) | (2u << 30); // addr[56:32] | type=2
    v8i g1;
    g1[0] = 0x00010000;                                   // wg_mask=0, data_size=1 (2 bytes)
    g1[1] = (int)((tile_w & 0xffffu) << 16);              // tensor_dim0 lo16 @ [63:48]
    g1[2] = (int)(((tile_w >> 16) & 0xffffu) | ((tile_h & 0xffffu) << 16)); // dim0 hi | dim1 lo
    g1[3] = (int)(((tile_h >> 16) & 0xffffu) | ((tile_w & 0xffffu) << 16)); // dim1 hi | tile_dim0
    g1[4] = (int)(tile_h & 0xffffu);                      // tile_dim1 (tile_dim2 = 0)
    g1[5] = (int)(unsigned)(row_stride & 0xffffffffu);    // tensor_dim0_stride lo32
    g1[6] = (int)(unsigned)((row_stride >> 32) & 0xffffu);// stride hi16, dim1_stride lo = 0
    g1[7] = 0;
    const v4i z4 = {0, 0, 0, 0};                          // groups 2/3 unused (2D tensor)
    const v8i z8 = {0, 0, 0, 0, 0, 0, 0, 0};
    __builtin_amdgcn_tensor_load_to_lds(g0, g1, z4, z4, z8, 0);
}

// ---- WMMA micro-step: one 32-deep K chunk, 2 M-subtiles x 1 N-subtile per wave ----
// LDS layouts per CDNA5 ISA 7.12.2 fragment maps (lda/ldb in u32 units).
__device__ __forceinline__ void wmma_step(const unsigned* lAu, int lda,
                                          const unsigned* lBu, int ldb,
                                          int wm, int wn, int lane,
                                          v8f& acc0, v8f& acc1)
{
    const int r = lane & 15, half = lane >> 4;
    union { v16bf v; unsigned u[8]; } a0, a1, bf;
    const int col = wn * 16 + r;
#pragma unroll
    for (int j = 0; j < 8; ++j)
        bf.u[j] = lBu[col * ldb + half * 4 + (j & 3) + (j >> 2) * 8]; // B: K pair map (same as A)
    const int row0 = wm * 32 + r;
#pragma unroll
    for (int j = 0; j < 8; ++j) {
        const int idx = half * 4 + (j & 3) + (j >> 2) * 8;  // A: K pair map
        a0.u[j] = lAu[row0 * lda + idx];
        a1.u[j] = lAu[(row0 + 16) * lda + idx];
    }
    acc0 = __builtin_amdgcn_wmma_f32_16x16x32_bf16(false, a0.v, false, bf.v, (short)0, acc0, false, false);
    acc1 = __builtin_amdgcn_wmma_f32_16x16x32_bf16(false, a1.v, false, bf.v, (short)0, acc1, false, false);
}

// ======================= Kernel 0: fp32 -> bf16 weight conversion =======================
__global__ void __launch_bounds__(256)
cvt_bf16(const float* __restrict__ s, __bf16* __restrict__ d, int n)
{
    const int i = (blockIdx.x * 256 + threadIdx.x) * 4;
    if (i + 3 < n) {
        const float4 f = *(const float4*)(s + i);
        d[i + 0] = (__bf16)f.x; d[i + 1] = (__bf16)f.y;
        d[i + 2] = (__bf16)f.z; d[i + 3] = (__bf16)f.w;
    }
}

// ======================= Kernel 1: patchify conv + BN + GELU =======================
// M=2048, N=256, K=16384. A: gathered 8x8 windows from x (fp32->bf16); B: TDM strided tile.
__global__ void __launch_bounds__(256)
gemm_patchify(const float* __restrict__ x, const __bf16* __restrict__ Wp,
              const float* __restrict__ bp, const float* __restrict__ gamma,
              const float* __restrict__ beta, const float* __restrict__ mean,
              const float* __restrict__ var, __bf16* __restrict__ pf)
{
    __shared__ __bf16 lA[64 * 32];
    __shared__ __bf16 lB[64 * 32];
    const int t = threadIdx.x;
    const int lane = t & 31, wave = t >> 5;
    const int wm = wave >> 2, wn = wave & 3;
    const int m0 = blockIdx.x * 64, n0 = blockIdx.y * 64;

    const int ml = t >> 2, part = t & 3;
    const int m = m0 + ml;
    const int b = m >> 10, p = m & 1023;
    const int py = p >> 5, px = p & 31;
    const unsigned lB_off = lds_off(lB);

    v8f acc[2] = {};
    for (int kc = 0; kc < 16384; kc += 32) {
        if (wave == 0)  // TDM: 64 rows (n), 32 elems, row stride 16384 (EXEC-agnostic -> one wave)
            tdm_load_tile_bf16(Wp + (size_t)n0 * 16384 + kc, lB_off, 32, 64, 16384ull);
        const int k = kc + part * 8;          // kx = 0..7 within one (c, ky)
        const int c = k >> 6, ky = (k >> 3) & 7;
        const float* src = x + (((size_t)(b * 256 + c) * 256) + (py * 8 + ky)) * 256 + px * 8;
        __bf16* dst = lA + ml * 32 + part * 8;
#pragma unroll
        for (int i = 0; i < 8; ++i) dst[i] = (__bf16)src[i];
        if (wave == 0) __builtin_amdgcn_s_wait_tensorcnt(0);
        __syncthreads();
        wmma_step((const unsigned*)lA, 16, (const unsigned*)lB, 16, wm, wn, lane, acc[0], acc[1]);
        __syncthreads();
    }

    const int r = lane & 15, half = lane >> 4;
    const int n = n0 + wn * 16 + r;
    const float invs = gamma[n] * rsqrtf(var[n] + 1e-5f);
    const float mu = mean[n], bb = beta[n], bias = bp[n];
#pragma unroll
    for (int sub = 0; sub < 2; ++sub) {
#pragma unroll
        for (int v = 0; v < 8; ++v) {
            const int mg = m0 + wm * 32 + sub * 16 + half * 8 + v;
            float val = acc[sub][v] + bias;
            val = (val - mu) * invs + bb;
            val = 0.5f * val * (1.0f + erff(val * 0.70710678118654752f)); // exact GELU
            pf[(size_t)mg * 256 + n] = (__bf16)val;
        }
    }
}

// ======================= Kernel 2: q = 1x1 conv, scatter to (bn,p,qi,d) =======================
// M=131072, N=256, K=256. A: transpose-gather from NCHW x; B: full 64x256 panel via one TDM.
__global__ void __launch_bounds__(256)
gemm_q(const float* __restrict__ x, const __bf16* __restrict__ Wq,
       const float* __restrict__ bq, __bf16* __restrict__ qb)
{
    __shared__ __bf16 lA[64 * 32];
    __shared__ __bf16 lB[64 * 256];
    const int t = threadIdx.x;
    const int lane = t & 31, wave = t >> 5;
    const int wm = wave >> 2, wn = wave & 3;
    const int m0 = blockIdx.x * 64, n0 = blockIdx.y * 64;

    if (wave == 0) {
        tdm_load_tile_bf16(Wq + (size_t)n0 * 256, lds_off(lB), 256, 64, 256ull);
        __builtin_amdgcn_s_wait_tensorcnt(0);
    }

    v8f acc[2] = {};
    for (int kc = 0; kc < 256; kc += 32) {
#pragma unroll
        for (int pass = 0; pass < 8; ++pass) {            // x is [b][c][hw]; coalesce along hw
            const int cl = (t >> 6) + pass * 4;           // 0..31
            const int ml = t & 63;
            const int mm = m0 + ml;
            const int b = mm >> 16, hw = mm & 65535;
            lA[ml * 32 + cl] = (__bf16)x[(size_t)(b * 256 + kc + cl) * 65536 + hw];
        }
        __syncthreads();
        wmma_step((const unsigned*)lA, 16, (const unsigned*)lB + (kc >> 1), 128,
                  wm, wn, lane, acc[0], acc[1]);
        __syncthreads();
    }

    const int r = lane & 15, half = lane >> 4;
    const int o = n0 + wn * 16 + r;
    const int head = o >> 5, d = o & 31;
    const float bias = bq[o];
#pragma unroll
    for (int sub = 0; sub < 2; ++sub) {
#pragma unroll
        for (int v = 0; v < 8; ++v) {
            const int mg = m0 + wm * 32 + sub * 16 + half * 8 + v;
            const int b = mg >> 16, hw = mg & 65535;
            const int h = hw >> 8, w = hw & 255;
            const int bn = b * 8 + head;
            const int pp = (h >> 3) * 32 + (w >> 3);
            const int qi = (h & 7) * 8 + (w & 7);
            qb[(size_t)((bn * 1024 + pp) * 64 + qi) * 32 + d] = (__bf16)(acc[sub][v] + bias);
        }
    }
}

// ======================= Kernel 3: kv = 1x1 conv on pf -> k/v (bn,p,d) =======================
// M=2048, N=512, K=256. Full A and B panels via TDM, one barrier, 8 WMMA chunks from LDS.
__global__ void __launch_bounds__(256)
gemm_kv(const __bf16* __restrict__ pf, const __bf16* __restrict__ Wkv,
        const float* __restrict__ bkv, __bf16* __restrict__ kb, __bf16* __restrict__ vb)
{
    __shared__ __bf16 lA[64 * 256];
    __shared__ __bf16 lB[64 * 256];
    const int t = threadIdx.x;
    const int lane = t & 31, wave = t >> 5;
    const int wm = wave >> 2, wn = wave & 3;
    const int m0 = blockIdx.x * 64, n0 = blockIdx.y * 64;

    if (wave == 0) {
        tdm_load_tile_bf16(pf  + (size_t)m0 * 256, lds_off(lA), 256, 64, 256ull);
        tdm_load_tile_bf16(Wkv + (size_t)n0 * 256, lds_off(lB), 256, 64, 256ull);
        __builtin_amdgcn_s_wait_tensorcnt(0);
    }
    __syncthreads();

    v8f acc[2] = {};
#pragma unroll
    for (int kc = 0; kc < 8; ++kc)
        wmma_step((const unsigned*)lA + kc * 16, 128, (const unsigned*)lB + kc * 16, 128,
                  wm, wn, lane, acc[0], acc[1]);

    const int r = lane & 15, half = lane >> 4;
    const int j = n0 + wn * 16 + r;
    const float bias = bkv[j];
    const int headk = j >> 6;     // kv reshaped (B, 8 heads, 64, hh, ww)
    const int jj = j & 63;
#pragma unroll
    for (int sub = 0; sub < 2; ++sub) {
#pragma unroll
        for (int v = 0; v < 8; ++v) {
            const int mg = m0 + wm * 32 + sub * 16 + half * 8 + v;
            const int b = mg >> 10, p = mg & 1023;
            const int bn = b * 8 + headk;
            const __bf16 val = (__bf16)(acc[sub][v] + bias);
            if (jj < 32) kb[(size_t)(bn * 1024 + p) * 32 + jj] = val;
            else         vb[(size_t)(bn * 1024 + p) * 32 + (jj - 32)] = val;
        }
    }
}

// ======================= Kernel 4: windowed local attention =======================
// One 64-thread block per (bn, p). 9 neighbors incl. zero-padded OOB (they DO enter softmax).
__global__ void __launch_bounds__(64)
attn_kernel(const __bf16* __restrict__ qb, const __bf16* __restrict__ kb,
            const __bf16* __restrict__ vb, __bf16* __restrict__ ao)
{
    __shared__ float kn[9][32];
    __shared__ float vn[9][32];
    const int blk = blockIdx.x;
    const int bn = blk >> 10, p = blk & 1023;
    const int py = p >> 5, px = p & 31;
    const int t = threadIdx.x;

    for (int idx = t; idx < 288; idx += 64) {
        const int nb = idx >> 5, d = idx & 31;
        const int ny = py + nb / 3 - 1, nx = px + nb % 3 - 1;
        float kvv = 0.f, vvv = 0.f;
        if (ny >= 0 && ny < 32 && nx >= 0 && nx < 32) {
            const size_t base = (size_t)(bn * 1024 + ny * 32 + nx) * 32 + d;
            kvv = (float)kb[base];
            vvv = (float)vb[base];
        }
        kn[nb][d] = kvv;
        vn[nb][d] = vvv;
    }
    __syncthreads();

    const __bf16* qp = qb + (size_t)((bn * 1024 + p) * 64 + t) * 32;
    float qv[32];
#pragma unroll
    for (int d = 0; d < 32; ++d) qv[d] = (float)qp[d];

    const float scale = 0.17677669529663687f; // 32^-0.5
    float lg[9];
#pragma unroll
    for (int nb = 0; nb < 9; ++nb) {
        float s = 0.f;
#pragma unroll
        for (int d = 0; d < 32; ++d) s += qv[d] * kn[nb][d];
        lg[nb] = s * scale;
    }
    float mx = lg[0];
#pragma unroll
    for (int nb = 1; nb < 9; ++nb) mx = fmaxf(mx, lg[nb]);
    float ssum = 0.f;
#pragma unroll
    for (int nb = 0; nb < 9; ++nb) { lg[nb] = __expf(lg[nb] - mx); ssum += lg[nb]; }
    const float inv = 1.f / ssum;

    float oacc[32] = {};
#pragma unroll
    for (int nb = 0; nb < 9; ++nb) {
        const float w = lg[nb] * inv;
#pragma unroll
        for (int d = 0; d < 32; ++d) oacc[d] += w * vn[nb][d];
    }

    const int b = bn >> 3, head = bn & 7;
    const int h = py * 8 + (t >> 3), w = px * 8 + (t & 7);
    __bf16* op = ao + (size_t)(b * 65536 + h * 256 + w) * 256 + head * 32;
#pragma unroll
    for (int d = 0; d < 32; ++d) op[d] = (__bf16)oacc[d];
}

// ======================= Kernel 5: output 1x1 conv (Wo) -> fp32 NCHW =======================
// M=131072, N=256, K=256. Full A and B panels via TDM.
__global__ void __launch_bounds__(256)
gemm_out(const __bf16* __restrict__ ao, const __bf16* __restrict__ Wo,
         const float* __restrict__ bo, float* __restrict__ out)
{
    __shared__ __bf16 lA[64 * 256];
    __shared__ __bf16 lB[64 * 256];
    const int t = threadIdx.x;
    const int lane = t & 31, wave = t >> 5;
    const int wm = wave >> 2, wn = wave & 3;
    const int m0 = blockIdx.x * 64, n0 = blockIdx.y * 64;

    if (wave == 0) {
        tdm_load_tile_bf16(ao + (size_t)m0 * 256, lds_off(lA), 256, 64, 256ull);
        tdm_load_tile_bf16(Wo + (size_t)n0 * 256, lds_off(lB), 256, 64, 256ull);
        __builtin_amdgcn_s_wait_tensorcnt(0);
    }
    __syncthreads();

    v8f acc[2] = {};
#pragma unroll
    for (int kc = 0; kc < 8; ++kc)
        wmma_step((const unsigned*)lA + kc * 16, 128, (const unsigned*)lB + kc * 16, 128,
                  wm, wn, lane, acc[0], acc[1]);

    const int r = lane & 15, half = lane >> 4;
    const int o = n0 + wn * 16 + r;
    const float bias = bo[o];
#pragma unroll
    for (int sub = 0; sub < 2; ++sub) {
#pragma unroll
        for (int v = 0; v < 8; ++v) {
            const int mg = m0 + wm * 32 + sub * 16 + half * 8 + v;
            const int b = mg >> 16, hw = mg & 65535;
            out[(size_t)(b * 256 + o) * 65536 + hw] = acc[sub][v] + bias;
        }
    }
}

// ======================= Launcher =======================
extern "C" void kernel_launch(void* const* d_in, const int* in_sizes, int n_in,
                              void* d_out, int out_size, void* d_ws, size_t ws_size,
                              hipStream_t stream)
{
    const float* x     = (const float*)d_in[0];
    const float* Wp    = (const float*)d_in[1];
    const float* bp    = (const float*)d_in[2];
    const float* gamma = (const float*)d_in[3];
    const float* beta  = (const float*)d_in[4];
    const float* mean  = (const float*)d_in[5];
    const float* var   = (const float*)d_in[6];
    const float* Wq    = (const float*)d_in[7];
    const float* bq    = (const float*)d_in[8];
    const float* Wkv   = (const float*)d_in[9];
    const float* bkv   = (const float*)d_in[10];
    const float* Wo    = (const float*)d_in[11];
    const float* bo    = (const float*)d_in[12];
    float* out = (float*)d_out;

    const size_t MB = (size_t)1 << 20;
    char* ws = (char*)d_ws;
    __bf16* pf   = (__bf16*)(ws + 0 * MB);    //  1 MB: 2048 x 256
    __bf16* kb   = (__bf16*)(ws + 1 * MB);    //  1 MB: 16 x 1024 x 32
    __bf16* vb   = (__bf16*)(ws + 2 * MB);    //  1 MB
    __bf16* qb   = (__bf16*)(ws + 3 * MB);    // 64 MB: 16 x 1024 x 64 x 32
    __bf16* ao   = (__bf16*)(ws + 67 * MB);   // 64 MB: 131072 x 256
    __bf16* WpB  = (__bf16*)(ws + 131 * MB);  //  8 MB: 256 x 16384
    __bf16* WqB  = (__bf16*)(ws + 140 * MB);  // 128 KB
    __bf16* WkvB = (__bf16*)(ws + 141 * MB);  // 256 KB
    __bf16* WoB  = (__bf16*)(ws + 142 * MB);  // 128 KB

    cvt_bf16<<<4096, 256, 0, stream>>>(Wp,  WpB,  256 * 16384);
    cvt_bf16<<<64,   256, 0, stream>>>(Wq,  WqB,  256 * 256);
    cvt_bf16<<<128,  256, 0, stream>>>(Wkv, WkvB, 512 * 256);
    cvt_bf16<<<64,   256, 0, stream>>>(Wo,  WoB,  256 * 256);

    gemm_patchify<<<dim3(32, 4),   256, 0, stream>>>(x, WpB, bp, gamma, beta, mean, var, pf);
    gemm_q       <<<dim3(2048, 4), 256, 0, stream>>>(x, WqB, bq, qb);
    gemm_kv      <<<dim3(32, 8),   256, 0, stream>>>(pf, WkvB, bkv, kb, vb);
    attn_kernel  <<<16384,          64, 0, stream>>>(qb, kb, vb, ao);
    gemm_out     <<<dim3(2048, 4), 256, 0, stream>>>(ao, WoB, bo, out);
}
